// SpatialAttentionModel_full_85014582657125
// MI455X (gfx1250) — matrix operand order repaired and checked by reading, not compiled, not run on hardware
//
#include <hip/hip_runtime.h>
#include <hip/hip_bf16.h>
#include <cstdint>

// ---------------------------------------------------------------------------
// Problem constants
// ---------------------------------------------------------------------------
#define BB 8
#define TT 12
#define NN 325
#define KH 8          // heads
#define DH 64         // head dim
#define DD 512        // model dim
#define BT 96         // B*T
#define MM 31200      // B*T*N  (== 1950 * 16, exact)
#define MT 1950       // M tiles of 16
#define NP 352        // N padded to multiple of 32 (22*16, 11*32)
#define SROW 360      // LDS row stride (floats)

typedef __attribute__((ext_vector_type(16))) _Float16 v16h;
typedef __attribute__((ext_vector_type(8)))  float    v8f;

// ---------------------------------------------------------------------------
// Fragment loaders (per CDNA5 ISA 7.12.2 layouts, wave32)
//  A (16x32 f16): lane l -> row m=l&15 ; halfs 0..7 = A[m][kb..kb+7],
//                 halfs 8..15 = A[m][kb+16..kb+23], kb=(l>>4)*8
//  B (32x16 f16, loaded from transposed Bt[n][k]): lane l -> col n=l&15 ;
//                 halfs 0..7 = k kb..kb+7, halfs 8..15 = k kb+16..kb+23, kb=(l>>4)*8
//  C/D (16x16 f32): lane l -> col n=l&15 ; vgpr j -> row m = j + (l>>4)*8
// ---------------------------------------------------------------------------
__device__ inline v16h load_afrag_h(const _Float16* __restrict__ a, int ld) {
  int l = threadIdx.x & 31;
  int m = l & 15, kb = (l >> 4) * 8;
  const _Float16* r = a + (size_t)m * ld + kb;
  v16h out;
  *(float4*)&out       = *(const float4*)r;
  *((float4*)&out + 1) = *(const float4*)(r + 16);
  return out;
}

__device__ inline v16h load_afrag_f32(const float* __restrict__ a, int ld) {
  int l = threadIdx.x & 31;
  int m = l & 15, kb = (l >> 4) * 8;
  const float* r = a + (size_t)m * ld + kb;
  v16h out;
#pragma unroll
  for (int j = 0; j < 8; ++j) out[j]     = (_Float16)r[j];
#pragma unroll
  for (int j = 0; j < 8; ++j) out[8 + j] = (_Float16)r[16 + j];
  return out;
}

__device__ inline v16h load_afrag_lds(const float* __restrict__ s, int ld, int k0) {
  int l = threadIdx.x & 31;
  int m = l & 15, kb = k0 + (l >> 4) * 8;
  v16h out;
#pragma unroll
  for (int j = 0; j < 8; ++j) out[j]     = (_Float16)s[m * ld + kb + j];
#pragma unroll
  for (int j = 0; j < 8; ++j) out[8 + j] = (_Float16)s[m * ld + kb + 16 + j];
  return out;
}

__device__ inline v16h load_bfrag_h(const _Float16* __restrict__ bt, int ld) {
  int l = threadIdx.x & 31;
  int n = l & 15, kb = (l >> 4) * 8;
  const _Float16* r = bt + (size_t)n * ld + kb;
  v16h out;
  *(float4*)&out       = *(const float4*)r;
  *((float4*)&out + 1) = *(const float4*)(r + 16);
  return out;
}

#define WMMA_F16(a, b, c) \
  __builtin_amdgcn_wmma_f32_16x16x32_f16(false, (a), false, (b), (short)0, (c), false, false)

// ---------------------------------------------------------------------------
// Utility kernels
// ---------------------------------------------------------------------------
__global__ void zero32_kernel(uint32_t* __restrict__ p, size_t n) {
  size_t i  = (size_t)blockIdx.x * blockDim.x + threadIdx.x;
  size_t st = (size_t)gridDim.x * blockDim.x;
  for (; i < n; i += st) p[i] = 0u;
}

// W [kin][nout] f32 row-major  ->  Wt [nout][kin] f16
__global__ void wtrans_kernel(const float* __restrict__ W, _Float16* __restrict__ Wt,
                              int kin, int nout) {
  size_t total = (size_t)kin * nout;
  size_t i  = (size_t)blockIdx.x * blockDim.x + threadIdx.x;
  size_t st = (size_t)gridDim.x * blockDim.x;
  for (; i < total; i += st) {
    int k = (int)(i / nout);
    int n = (int)(i - (size_t)k * nout);
    Wt[(size_t)n * kin + k] = (_Float16)W[i];
  }
}

// ---------------------------------------------------------------------------
// QKV projection: out = relu([X|STE] @ W + b), scattered to per-head layout.
// grid (1950, 8), 1 wave / block. vstore=0: [bt][h][n][e] ; 1: [bt][h][e][n]
// ---------------------------------------------------------------------------
__global__ __launch_bounds__(32) void qkv_gemm_kernel(
    const float* __restrict__ X, const float* __restrict__ STE,
    const _Float16* __restrict__ Wt /*[512][1024]*/,
    const float* __restrict__ bias, _Float16* __restrict__ out, int vstore) {
  const int mt = blockIdx.x, cg = blockIdx.y;
  const int l  = threadIdx.x & 31;
  const int rowbase = mt * 16;
  v8f acc[4] = {};
  for (int kk = 0; kk < 32; ++kk) {
    const int ks = kk * 32;
    const float* Ab = (ks < DD) ? (X + (size_t)rowbase * DD + ks)
                                : (STE + (size_t)rowbase * DD + (ks - DD));
    v16h a = load_afrag_f32(Ab, DD);
#pragma unroll
    for (int ct = 0; ct < 4; ++ct) {
      v16h b = load_bfrag_h(Wt + (size_t)(cg * 64 + ct * 16) * (2 * DD) + ks, 2 * DD);
      acc[ct] = WMMA_F16(a, b, acc[ct]);
    }
  }
#pragma unroll
  for (int ct = 0; ct < 4; ++ct) {
    const int col = cg * 64 + ct * 16 + (l & 15);
    const int h = col >> 6, e = col & 63;
    const float bv = bias[col];
#pragma unroll
    for (int j = 0; j < 8; ++j) {
      const int R  = rowbase + j + ((l >> 4) * 8);
      const int bt = R / NN, n = R - bt * NN;
      float v = acc[ct][j] + bv;
      v = v > 0.f ? v : 0.f;
      size_t idx = vstore ? ((size_t)((bt * KH + h) * DH + e) * NP + n)
                          : ((size_t)((bt * KH + h) * NP + n) * DH + e);
      out[idx] = (_Float16)v;
    }
  }
}

// ---------------------------------------------------------------------------
// Fused attention: scores (WMMA) -> LDS -> mask/scale/softmax -> attn f32 out
// and ctx = P @ V (WMMA, A from LDS).  grid (21, 8 heads, 96 bt), 1 wave.
// ---------------------------------------------------------------------------
__global__ __launch_bounds__(32) void attn_kernel(
    const _Float16* __restrict__ qh, const _Float16* __restrict__ kh,
    const _Float16* __restrict__ vth,
    float* __restrict__ attn_out, _Float16* __restrict__ ctx) {
  __shared__ float s[16 * SROW];
  const int mt = blockIdx.x, h = blockIdx.y, bt = blockIdx.z;
  const int l  = threadIdx.x & 31;
  const size_t head = (size_t)(bt * KH + h) * NP * DH;

  // ---- phase 1: S = q @ k^T  (16 x 352 strip) ----
  v16h a0 = load_afrag_h(qh + head + (size_t)(mt * 16) * DH + 0,  DH);
  v16h a1 = load_afrag_h(qh + head + (size_t)(mt * 16) * DH + 32, DH);
  for (int ct = 0; ct < 22; ++ct) {
    v16h b0 = load_bfrag_h(kh + head + (size_t)(ct * 16) * DH + 0,  DH);
    v16h b1 = load_bfrag_h(kh + head + (size_t)(ct * 16) * DH + 32, DH);
    v8f acc = {};
    acc = WMMA_F16(a0, b0, acc);
    acc = WMMA_F16(a1, b1, acc);
    const int cn = ct * 16 + (l & 15);
#pragma unroll
    for (int j = 0; j < 8; ++j) s[(j + ((l >> 4) * 8)) * SROW + cn] = acc[j];
  }
  __syncthreads();

  // ---- phase 2: causal mask, scale 1/sqrt(64), softmax; emit attn f32 ----
  const int b = bt / TT, t = bt - b * TT;
  for (int r = 0; r < 16; ++r) {
    const int n = mt * 16 + r;
    float vals[11];
    float mx = -3.0e38f;
#pragma unroll
    for (int i = 0; i < 11; ++i) {
      const int c = l + i * 32;
      const float v = s[r * SROW + c] * 0.125f;
      vals[i] = (c <= n) ? v : -3.0e38f;
      mx = fmaxf(mx, vals[i]);
    }
#pragma unroll
    for (int o = 16; o > 0; o >>= 1) mx = fmaxf(mx, __shfl_xor(mx, o, 32));
    float sum = 0.f;
#pragma unroll
    for (int i = 0; i < 11; ++i) {
      const int c = l + i * 32;
      const float e = (c <= n) ? __expf(vals[i] - mx) : 0.f;
      vals[i] = e;
      sum += e;
    }
#pragma unroll
    for (int o = 16; o > 0; o >>= 1) sum += __shfl_xor(sum, o, 32);
    const float inv = 1.f / sum;
#pragma unroll
    for (int i = 0; i < 11; ++i) {
      const int c = l + i * 32;
      const float p = vals[i] * inv;
      s[r * SROW + c] = p;
      if (n < NN && c < NN)
        attn_out[((((size_t)h * BB + b) * TT + t) * NN + n) * NN + c] = p;
    }
  }
  __syncthreads();

  // ---- phase 3: ctx = P @ V  (A from LDS, B from transposed V) ----
  v8f acc[4] = {};
  for (int kk = 0; kk < 11; ++kk) {
    v16h a = load_afrag_lds(s, SROW, kk * 32);
#pragma unroll
    for (int ct = 0; ct < 4; ++ct) {
      v16h bf = load_bfrag_h(vth + head + (size_t)(ct * 16) * NP + kk * 32, NP);
      acc[ct] = WMMA_F16(a, bf, acc[ct]);
    }
  }
#pragma unroll
  for (int ct = 0; ct < 4; ++ct) {
    const int e = ct * 16 + (l & 15);
#pragma unroll
    for (int j = 0; j < 8; ++j) {
      const int n = mt * 16 + j + ((l >> 4) * 8);
      if (n < NN)
        ctx[((size_t)bt * NN + n) * DD + h * DH + e] = (_Float16)acc[ct][j];
    }
  }
}

// ---------------------------------------------------------------------------
// FFN GEMM: C = act(A @ Wt^T + b); f16 out (hidden) or f32 out (final).
// grid (1950, 8), 1 wave / block.
// ---------------------------------------------------------------------------
__global__ __launch_bounds__(32) void ffn_gemm_kernel(
    const _Float16* __restrict__ A /*[M][512]*/,
    const _Float16* __restrict__ Wt /*[512][512]*/,
    const float* __restrict__ bias,
    _Float16* __restrict__ outh, float* __restrict__ outf, int do_relu) {
  const int mt = blockIdx.x, cg = blockIdx.y;
  const int l  = threadIdx.x & 31;
  const int rowbase = mt * 16;
  v8f acc[4] = {};
  for (int kk = 0; kk < 16; ++kk) {
    v16h a = load_afrag_h(A + (size_t)rowbase * DD + kk * 32, DD);
#pragma unroll
    for (int ct = 0; ct < 4; ++ct) {
      v16h b = load_bfrag_h(Wt + (size_t)(cg * 64 + ct * 16) * DD + kk * 32, DD);
      acc[ct] = WMMA_F16(a, b, acc[ct]);
    }
  }
#pragma unroll
  for (int ct = 0; ct < 4; ++ct) {
    const int col = cg * 64 + ct * 16 + (l & 15);
    const float bv = bias[col];
#pragma unroll
    for (int j = 0; j < 8; ++j) {
      const int R = rowbase + j + ((l >> 4) * 8);
      float v = acc[ct][j] + bv;
      if (do_relu) v = v > 0.f ? v : 0.f;
      if (outh) outh[(size_t)R * DD + col] = (_Float16)v;
      else      outf[(size_t)R * DD + col] = v;
    }
  }
}

// ---------------------------------------------------------------------------
// Host-side launcher
// ---------------------------------------------------------------------------
extern "C" void kernel_launch(void* const* d_in, const int* /*in_sizes*/, int /*n_in*/,
                              void* d_out, int /*out_size*/, void* d_ws, size_t /*ws_size*/,
                              hipStream_t stream) {
  const float* X   = (const float*)d_in[0];
  const float* STE = (const float*)d_in[1];
  const float* Wq  = (const float*)d_in[2];
  const float* bq  = (const float*)d_in[3];
  const float* Wk  = (const float*)d_in[4];
  const float* bk  = (const float*)d_in[5];
  const float* Wv  = (const float*)d_in[6];
  const float* bv  = (const float*)d_in[7];
  const float* W1  = (const float*)d_in[8];
  const float* b1  = (const float*)d_in[9];
  const float* W2  = (const float*)d_in[10];
  const float* b2  = (const float*)d_in[11];

  float* out_f    = (float*)d_out;                           // [B,T,N,D]
  float* attn_f   = out_f + (size_t)MM * DD;                 // [K*B,T,N,N]

  // workspace carve-up (256B aligned)
  uint8_t* ws = (uint8_t*)d_ws;
  size_t off = 0;
  auto carve = [&](size_t bytes) -> uint8_t* {
    uint8_t* p = ws + off;
    off = (off + bytes + 255) & ~(size_t)255;
    return p;
  };
  const size_t QKV_HALFS = (size_t)BT * KH * NP * DH;        // 17,301,504
  _Float16* qh  = (_Float16*)carve(QKV_HALFS * 2);
  _Float16* kh  = (_Float16*)carve(QKV_HALFS * 2);
  _Float16* vth = (_Float16*)carve(QKV_HALFS * 2);
  _Float16* ctx = (_Float16*)carve((size_t)MM * DD * 2);
  _Float16* WqT = (_Float16*)carve((size_t)DD * 2 * DD * 2);
  _Float16* WkT = (_Float16*)carve((size_t)DD * 2 * DD * 2);
  _Float16* WvT = (_Float16*)carve((size_t)DD * 2 * DD * 2);
  _Float16* W1T = (_Float16*)carve((size_t)DD * DD * 2);
  _Float16* W2T = (_Float16*)carve((size_t)DD * DD * 2);
  _Float16* hid = qh;  // alias: qh dead after attention kernel

  // 1) zero padded Q/K/V buffers (padding rows must contribute exactly 0)
  zero32_kernel<<<2048, 256, 0, stream>>>((uint32_t*)qh,  QKV_HALFS / 2);
  zero32_kernel<<<2048, 256, 0, stream>>>((uint32_t*)kh,  QKV_HALFS / 2);
  zero32_kernel<<<2048, 256, 0, stream>>>((uint32_t*)vth, QKV_HALFS / 2);

  // 2) transpose + f16-convert weights
  wtrans_kernel<<<2048, 256, 0, stream>>>(Wq, WqT, 2 * DD, DD);
  wtrans_kernel<<<2048, 256, 0, stream>>>(Wk, WkT, 2 * DD, DD);
  wtrans_kernel<<<2048, 256, 0, stream>>>(Wv, WvT, 2 * DD, DD);
  wtrans_kernel<<<2048, 256, 0, stream>>>(W1, W1T, DD, DD);
  wtrans_kernel<<<2048, 256, 0, stream>>>(W2, W2T, DD, DD);

  // 3) Q/K/V projections (relu), scattered to per-head (padded) layouts
  dim3 gemm_grid(MT, 8);
  qkv_gemm_kernel<<<gemm_grid, 32, 0, stream>>>(X, STE, WqT, bq, qh,  0);
  qkv_gemm_kernel<<<gemm_grid, 32, 0, stream>>>(X, STE, WkT, bk, kh,  0);
  qkv_gemm_kernel<<<gemm_grid, 32, 0, stream>>>(X, STE, WvT, bv, vth, 1);

  // 4) fused attention: scores -> softmax -> attn output + ctx
  dim3 attn_grid(22 - 1 /*21 row tiles cover 325*/, KH, BT);
  attn_kernel<<<attn_grid, 32, 0, stream>>>(qh, kh, vth, attn_f, ctx);

  // 5) FFN
  ffn_gemm_kernel<<<gemm_grid, 32, 0, stream>>>(ctx, W1T, b1, hid, nullptr, 1);
  ffn_gemm_kernel<<<gemm_grid, 32, 0, stream>>>(hid, W2T, b2, nullptr, out_f, 0);
}
